// Net_87694642250250
// MI455X (gfx1250) — matrix operand order repaired and checked by reading, not compiled, not run on hardware
//
#include <hip/hip_runtime.h>
#include <hip/hip_bf16.h>

// CDNA5 / gfx1250, wave32 only.
typedef __attribute__((ext_vector_type(16))) __bf16 v16bf;
typedef __attribute__((ext_vector_type(8)))  float  v8f;

// ---------------------------------------------------------------- utilities
__global__ void k_fill(float* __restrict__ p, float v, long n) {
    long i = (long)blockIdx.x * blockDim.x + threadIdx.x;
    if (i < n) p[i] = v;
}

__global__ void k_degcount(const int* __restrict__ dst, float* __restrict__ deg, int E) {
    int i = blockIdx.x * blockDim.x + threadIdx.x;
    if (i < E) atomicAdd(&deg[dst[i]], 1.0f);
}

__global__ void k_rsqrt(float* __restrict__ deg, int n) {
    int i = blockIdx.x * blockDim.x + threadIdx.x;
    if (i < n) { float d = deg[i]; deg[i] = d > 0.0f ? rsqrtf(d) : 0.0f; }
}

// ---------------------------------------------------------------- WMMA GEMM
// Y[N x FOUT] = X[N x FIN] @ W[FIN x FOUT], one 16x16 tile per wave32,
// bf16 inputs (converted on the fly), f32 accumulate via
// v_wmma_f32_16x16x32_bf16.  EXEC is uniform per wave (tile-level early out),
// so WMMA's all-lanes-active requirement holds.
template<int FIN, int FOUT>
__global__ void k_gemm_wmma(const float* __restrict__ X,
                            const float* __restrict__ W,
                            float* __restrict__ Y,
                            int rowTiles, int nrows) {
    static_assert(FIN % 32 == 0 && FOUT % 16 == 0, "tile shapes");
    const int wv   = (int)(((long)blockIdx.x * blockDim.x + threadIdx.x) >> 5);
    const int lane = threadIdx.x & 31;
    constexpr int CT = FOUT / 16;
    const int rt = wv / CT;
    const int ct = wv % CT;
    if (rt >= rowTiles) return;            // uniform per wave

    const int m    = lane & 15;            // row within tile (A) / col (B,D)
    const int half = lane >> 4;            // lane group 0/1
    int arowi = rt * 16 + m;
    if (arowi >= nrows) arowi = nrows - 1; // tail clamp (harmless duplicate)
    const float* __restrict__ arow = X + (size_t)arowi * FIN;
    const int n = ct * 16 + m;             // output column

    v8f acc = {};
    #pragma unroll
    for (int k0 = 0; k0 < FIN; k0 += 32) {
        // A tile 16x32 bf16: lane layout per ISA 7.12.2 (16-bit A-matrix).
        // j=0..7 -> K = k0 + 8*half + j ; j=8..15 -> K = k0 + 16 + 8*half + (j-8)
        float4 a0 = *(const float4*)(arow + k0 + 8 * half);
        float4 a1 = *(const float4*)(arow + k0 + 8 * half + 4);
        float4 a2 = *(const float4*)(arow + k0 + 16 + 8 * half);
        float4 a3 = *(const float4*)(arow + k0 + 20 + 8 * half);
        v16bf A;
        A[0]  = (__bf16)a0.x; A[1]  = (__bf16)a0.y; A[2]  = (__bf16)a0.z; A[3]  = (__bf16)a0.w;
        A[4]  = (__bf16)a1.x; A[5]  = (__bf16)a1.y; A[6]  = (__bf16)a1.z; A[7]  = (__bf16)a1.w;
        A[8]  = (__bf16)a2.x; A[9]  = (__bf16)a2.y; A[10] = (__bf16)a2.z; A[11] = (__bf16)a2.w;
        A[12] = (__bf16)a3.x; A[13] = (__bf16)a3.y; A[14] = (__bf16)a3.z; A[15] = (__bf16)a3.w;
        // B tile 32x16 bf16: element j -> K = k0 + 8*half + j + (j & 8), col n.
        v16bf B;
        #pragma unroll
        for (int j = 0; j < 16; ++j)
            B[j] = (__bf16)W[(size_t)(k0 + 8 * half + j + (j & 8)) * FOUT + n];
        acc = __builtin_amdgcn_wmma_f32_16x16x32_bf16(
                  false, A, false, B, (short)0, acc, false, false);
    }
    // D layout: element r -> row rt*16 + r + 8*half, col n.
    const int rbase = rt * 16 + 8 * half;
    float* __restrict__ yout = Y + (size_t)rbase * FOUT + n;
    #pragma unroll
    for (int r = 0; r < 8; ++r)
        if (rbase + r < nrows) yout[(size_t)r * FOUT] = acc[r];
}

// ---------------------------------------------------------------- scatter-add
// One wave32 per edge: out[dst] += dinv[src]*dinv[dst] * xw[src]  (F floats)
template<int F>
__global__ void k_scatter(const int* __restrict__ src, const int* __restrict__ dst,
                          const float* __restrict__ dinv, const float* __restrict__ xw,
                          float* __restrict__ out, int E) {
    long gid = (long)blockIdx.x * blockDim.x + threadIdx.x;
    int e    = (int)(gid >> 5);
    int lane = (int)(gid & 31);
    if (e >= E) return;
    const int s = src[e], d = dst[e];
    const float norm = dinv[s] * dinv[d];
    constexpr int V = F / 32;                      // 4 (F=128) or 2 (F=64)
    const float* __restrict__ xs = xw + (size_t)s * F + lane * V;
    float* __restrict__       od = out + (size_t)d * F + lane * V;
    if constexpr (V == 4) {
        float4 v = *(const float4*)xs;
        atomicAdd(od + 0, norm * v.x); atomicAdd(od + 1, norm * v.y);
        atomicAdd(od + 2, norm * v.z); atomicAdd(od + 3, norm * v.w);
    } else {
        float2 v = *(const float2*)xs;
        atomicAdd(od + 0, norm * v.x); atomicAdd(od + 1, norm * v.y);
    }
}

// epilogue: self-loop term + bias (+ ReLU)
template<int F, bool RELU>
__global__ void k_finish(const float* __restrict__ xw, const float* __restrict__ dinv,
                         const float* __restrict__ b, float* __restrict__ out, int n) {
    long gid = (long)blockIdx.x * blockDim.x + threadIdx.x;
    if (gid >= (long)n * F) return;
    const int node = (int)(gid / F);
    const int f    = (int)(gid % F);
    const float di = dinv[node];
    float v = out[gid] + di * di * xw[gid] + b[f];
    out[gid] = RELU ? fmaxf(v, 0.0f) : v;
}

// ---------------------------------------------------------------- scores
// One wave32 per edge, 64-dim dot with shfl_xor wave reduction.
__global__ void k_scores(const int* __restrict__ pos, const int* __restrict__ neg,
                         const float* __restrict__ h, float* __restrict__ out, int EP) {
    long gid = (long)blockIdx.x * blockDim.x + threadIdx.x;
    int e    = (int)(gid >> 5);
    int lane = (int)(gid & 31);
    if (e >= 2 * EP) return;
    int j, i;
    if (e < EP) { j = pos[e];      i = pos[EP + e]; }
    else        { j = neg[e - EP]; i = neg[e];      }
    float2 a = *(const float2*)(h + (size_t)i * 64 + lane * 2);
    float2 c = *(const float2*)(h + (size_t)j * 64 + lane * 2);
    float p = a.x * c.x + a.y * c.y;
    #pragma unroll
    for (int off = 16; off; off >>= 1) p += __shfl_xor(p, off, 32);
    if (lane == 0) out[e] = p;
}

// ---------------------------------------------------------------- driver
extern "C" void kernel_launch(void* const* d_in, const int* in_sizes, int n_in,
                              void* d_out, int out_size, void* d_ws, size_t ws_size,
                              hipStream_t stream) {
    const float* x   = (const float*)d_in[0];
    const int*   tei = (const int*)d_in[1];
    const int*   pei = (const int*)d_in[2];
    const int*   nei = (const int*)d_in[3];
    const float* W1  = (const float*)d_in[4];
    const float* b1  = (const float*)d_in[5];
    const float* W2  = (const float*)d_in[6];
    const float* b2  = (const float*)d_in[7];
    const float* W3  = (const float*)d_in[8];
    const float* b3  = (const float*)d_in[9];

    constexpr int F0 = 128, F1 = 128, F2 = 64, F3 = 64;
    const int N  = in_sizes[0] / F0;   // 100000
    const int E  = in_sizes[1] / 2;    // 1.6M real edges (self-loops handled analytically)
    const int EP = in_sizes[2] / 2;    // 500k
    const int* srcp = tei;
    const int* dstp = tei + E;

    float* scores = (float*)d_out;             // [2*EP]
    float* embed  = (float*)d_out + 2 * EP;    // [N*F2]

    // workspace: dinv[N] | bufXW[N*128] | bufH[N*128]   (~103 MB)
    float* dinv  = (float*)d_ws;
    float* bufXW = dinv + (((size_t)N + 255) & ~(size_t)255);
    float* bufH  = bufXW + (size_t)N * 128;

    auto cdiv = [](long a, long b) { return (int)((a + b - 1) / b); };
    const int rowTiles = (N + 15) / 16;

    // degrees -> dinv (init 1.0 accounts for the self-loop)
    k_fill<<<cdiv(N, 256), 256, 0, stream>>>(dinv, 1.0f, N);
    k_degcount<<<cdiv(E, 256), 256, 0, stream>>>(dstp, dinv, E);
    k_rsqrt<<<cdiv(N, 256), 256, 0, stream>>>(dinv, N);

    // layer 1: 128 -> 128, ReLU
    {
        long waves = (long)rowTiles * (F1 / 16);
        k_gemm_wmma<128, 128><<<cdiv(waves * 32, 256), 256, 0, stream>>>(x, W1, bufXW, rowTiles, N);
        k_fill<<<cdiv((long)N * F1, 256), 256, 0, stream>>>(bufH, 0.0f, (long)N * F1);
        k_scatter<128><<<cdiv((long)E * 32, 256), 256, 0, stream>>>(srcp, dstp, dinv, bufXW, bufH, E);
        k_finish<128, true><<<cdiv((long)N * F1, 256), 256, 0, stream>>>(bufXW, dinv, b1, bufH, N);
    }
    // layer 2: 128 -> 64, ReLU -> embed (direct into d_out slice)
    {
        long waves = (long)rowTiles * (F2 / 16);
        k_gemm_wmma<128, 64><<<cdiv(waves * 32, 256), 256, 0, stream>>>(bufH, W2, bufXW, rowTiles, N);
        k_fill<<<cdiv((long)N * F2, 256), 256, 0, stream>>>(embed, 0.0f, (long)N * F2);
        k_scatter<64><<<cdiv((long)E * 32, 256), 256, 0, stream>>>(srcp, dstp, dinv, bufXW, embed, E);
        k_finish<64, true><<<cdiv((long)N * F2, 256), 256, 0, stream>>>(bufXW, dinv, b2, embed, N);
    }
    // layer 3: 64 -> 64, no ReLU -> h3 in bufH
    {
        long waves = (long)rowTiles * (F3 / 16);
        k_gemm_wmma<64, 64><<<cdiv(waves * 32, 256), 256, 0, stream>>>(embed, W3, bufXW, rowTiles, N);
        k_fill<<<cdiv((long)N * F3, 256), 256, 0, stream>>>(bufH, 0.0f, (long)N * F3);
        k_scatter<64><<<cdiv((long)E * 32, 256), 256, 0, stream>>>(srcp, dstp, dinv, bufXW, bufH, E);
        k_finish<64, false><<<cdiv((long)N * F3, 256), 256, 0, stream>>>(bufXW, dinv, b3, bufH, N);
    }
    // edge scores
    k_scores<<<cdiv((long)2 * EP * 32, 256), 256, 0, stream>>>(pei, nei, bufH, scores, EP);
}
